// ContrastiveLoss_56212531970333
// MI455X (gfx1250) — matrix operand order repaired and checked by reading, not compiled, not run on hardware
//
#include <hip/hip_runtime.h>

typedef __attribute__((ext_vector_type(16))) _Float16 v16h;
typedef __attribute__((ext_vector_type(8)))  _Float16 v8h;
typedef __attribute__((ext_vector_type(8)))  float    v8f;

#define NROW 4096
#define DDIM 768
#define INV_TEMP 20.0f
#define CEPS 1e-8f
#define NCHUNK 8                       // column chunks (4096/8 = 512 cols = 32 tiles each)
#define TILES_PER_CHUNK (NROW / 16 / NCHUNK)
#define KSTEPS (DDIM / 32)             // 24 WMMA K-steps

#define AS3 __attribute__((address_space(3)))
#define MKB(lo, hi) __builtin_shufflevector(lo, hi, 0,1,2,3,4,5,6,7,8,9,10,11,12,13,14,15)

// ---------------- Kernel 1: L2-normalize rows of both features -> f16 ----------------
__global__ void norm_rows_kernel(const float* __restrict__ f1, const float* __restrict__ f2,
                                 _Float16* __restrict__ n1, _Float16* __restrict__ n2) {
    int row = blockIdx.x;
    const float* src;
    _Float16* dst;
    if (row < NROW) { src = f1 + (size_t)row * DDIM;          dst = n1 + (size_t)row * DDIM; }
    else            { src = f2 + (size_t)(row - NROW) * DDIM; dst = n2 + (size_t)(row - NROW) * DDIM; }

    int tid = threadIdx.x;                 // 256 threads, 3 elements each
    float x0 = src[tid];
    float x1 = src[tid + 256];
    float x2 = src[tid + 512];

    __shared__ float red[256];
    red[tid] = x0 * x0 + x1 * x1 + x2 * x2;
    __syncthreads();
    for (int s = 128; s > 0; s >>= 1) {
        if (tid < s) red[tid] += red[tid + s];
        __syncthreads();
    }
    float scl = 1.0f / fmaxf(sqrtf(red[0]), CEPS);

    dst[tid]       = (_Float16)(x0 * scl);
    dst[tid + 256] = (_Float16)(x1 * scl);
    dst[tid + 512] = (_Float16)(x2 * scl);
}

// ------- Kernel 2: fused WMMA GEMM + partial sum-exp + diagonal capture -------
// Grid: (32 row-blocks, 8 column-chunks). Block: 256 threads = 8 waves = 128 rows.
// A fragments: 24 x v16h per wave, register-resident (loaded once from global).
// B column tiles: staged in LDS via global_load_async_to_lds_b128 (double-buffered,
// shared by all 8 waves). B fragments use explicit even/odd register buffers with
// no copies, loaded two K-steps ahead, so ds_load latency overlaps the WMMA pipe.
__global__ void __launch_bounds__(256) contrastive_gemm_kernel(
        const _Float16* __restrict__ n1, const _Float16* __restrict__ n2,
        float* __restrict__ partial, float* __restrict__ diag) {
    extern __shared__ _Float16 lds[];
    _Float16* ldsB0 = lds;                          // 24 KB
    _Float16* ldsB1 = lds + 16 * DDIM;              // 24 KB

    int tid  = threadIdx.x;
    int wave = tid >> 5;
    int lane = tid & 31;
    int half = lane >> 4;                  // 0: lanes 0-15, 1: lanes 16-31
    int l16  = lane & 15;
    int m0   = blockIdx.x * 128 + wave * 16;       // strip base row
    int jt0  = blockIdx.y * TILES_PER_CHUNK;       // first column tile of this chunk

    // ---- A fragments: load directly into registers with the WMMA lane layout ----
    // lane L: row = L%16; halves 0-15 = K[ka..ka+15],
    // ka = kk*32 + (L<16 ? 0 : 16)
    const int aoff = half * 16;
    const _Float16* Arow = n1 + (size_t)(m0 + l16) * DDIM + aoff;
    v16h afrag[KSTEPS];
#pragma unroll
    for (int kk = 0; kk < KSTEPS; ++kk) {
        v8h alo = *(const v8h*)(Arow + kk * 32);
        v8h ahi = *(const v8h*)(Arow + kk * 32 + 8);
        afrag[kk] = MKB(alo, ahi);
    }

    // ---- async B-tile copy: one tile = 24576 B = 256 threads * 6 * 16 B ----
    const char* n2c = (const char*)n2;
#define ISSUE_B_TILE(JT, DSTPTR)                                                   \
    do {                                                                           \
        const char* _src = n2c + (size_t)(JT) * (16 * DDIM * 2);                   \
        char* _dst = (char*)(DSTPTR);                                              \
        _Pragma("unroll")                                                          \
        for (int _i = 0; _i < 6; ++_i) {                                           \
            unsigned _off = (unsigned)tid * 16u + (unsigned)_i * 4096u;            \
            AS3 char* _lp = (AS3 char*)(_dst + _off);                              \
            unsigned long long _g = (unsigned long long)(_src + _off);             \
            asm volatile("global_load_async_to_lds_b128 %0, %1, off"               \
                         :: "v"(_lp), "v"(_g) : "memory");                         \
        }                                                                          \
    } while (0)

    ISSUE_B_TILE(jt0, ldsB0);              // prologue

    float acc[8] = {0.f, 0.f, 0.f, 0.f, 0.f, 0.f, 0.f, 0.f};
    float myDiag = 0.f;
    const int diagTile = m0 >> 4;          // global tile index holding this strip's diagonal
    const bool ownsDiag = (blockIdx.y == (unsigned)(diagTile / TILES_PER_CHUNK));

    for (int j = 0; j < TILES_PER_CHUNK; ++j) {
        int jt = jt0 + j;
        const _Float16* Bcur  = (j & 1) ? ldsB1 : ldsB0;
        _Float16*       Bnext = (j & 1) ? ldsB0 : ldsB1;

        if (j + 1 < TILES_PER_CHUNK) {
            ISSUE_B_TILE(jt + 1, Bnext);                      // 6 loads in flight
            asm volatile("s_wait_asynccnt 0x6" ::: "memory"); // tile jt complete
        } else {
            asm volatile("s_wait_asynccnt 0x0" ::: "memory");
        }
        __syncthreads();                   // tile jt visible to all waves

        const _Float16* Bt = Bcur + l16 * DDIM + half * 16;

        // ---- even/odd B fragment buffers, no copies, loads 2 K-steps ahead ----
        v8h b0lo = *(const v8h*)(Bt);
        v8h b0hi = *(const v8h*)(Bt + 8);
        v8h b1lo = *(const v8h*)(Bt + 32);
        v8h b1hi = *(const v8h*)(Bt + 40);
        v8f c = {};
#pragma unroll
        for (int kk = 0; kk < KSTEPS; kk += 2) {
            c = __builtin_amdgcn_wmma_f32_16x16x32_f16(false, afrag[kk], false,
                                                       MKB(b0lo, b0hi),
                                                       (short)0, c, false, false);
            if (kk + 2 < KSTEPS) {
                b0lo = *(const v8h*)(Bt + (kk + 2) * 32);
                b0hi = *(const v8h*)(Bt + (kk + 2) * 32 + 8);
            }
            c = __builtin_amdgcn_wmma_f32_16x16x32_f16(false, afrag[kk + 1], false,
                                                       MKB(b1lo, b1hi),
                                                       (short)0, c, false, false);
            if (kk + 3 < KSTEPS) {
                b1lo = *(const v8h*)(Bt + (kk + 3) * 32);
                b1hi = *(const v8h*)(Bt + (kk + 3) * 32 + 8);
            }
        }

        // Capture diagonal logit: row r at lane r (VGPR r), row 8+r at lane 24+r
        if (jt == diagTile) {
#pragma unroll
            for (int r = 0; r < 8; ++r) {
                float v = c[r] * INV_TEMP;
                myDiag = (lane == r)      ? v : myDiag;
                myDiag = (lane == 24 + r) ? v : myDiag;
            }
        }
        // Partial sum-exp (cos<=1 -> logit<=20 -> exp<=4.9e8: no max shift needed)
#pragma unroll
        for (int r = 0; r < 8; ++r)
            acc[r] += __expf(c[r] * INV_TEMP);

        __syncthreads();                   // everyone done with Bcur before reuse
    }
#undef ISSUE_B_TILE

    // Row reduction: VGPR r holds row r on lanes 0-15, row 8+r on lanes 16-31
    float* pchunk = partial + (size_t)blockIdx.y * NROW;
#pragma unroll
    for (int r = 0; r < 8; ++r) {
        float s = acc[r];
        s += __shfl_xor(s, 1, 32);
        s += __shfl_xor(s, 2, 32);
        s += __shfl_xor(s, 4, 32);
        s += __shfl_xor(s, 8, 32);
        if (lane == 0)  pchunk[m0 + r]     = s;
        if (lane == 16) pchunk[m0 + 8 + r] = s;
        if (ownsDiag) {
            float dlo = __shfl(myDiag, r, 32);
            float dhi = __shfl(myDiag, 24 + r, 32);
            if (lane == 0)  diag[m0 + r]     = dlo;
            if (lane == 16) diag[m0 + 8 + r] = dhi;
        }
    }
}

// ---------------- Kernel 3: combine column-chunk partials -> per-row loss ----------------
__global__ void combine_kernel(const float* __restrict__ partial, const float* __restrict__ diag,
                               float* __restrict__ rowloss) {
    int i = blockIdx.x * 256 + threadIdx.x;
    float s = 0.f;
#pragma unroll
    for (int c = 0; c < NCHUNK; ++c) s += partial[(size_t)c * NROW + i];
    rowloss[i] = __logf(s) - diag[i];
}

// ---------------- Kernel 4: deterministic tree reduction -> scalar loss ----------------
__global__ void reduce_loss_kernel(const float* __restrict__ rowloss, float* __restrict__ out) {
    __shared__ float red[256];
    int tid = threadIdx.x;
    float s = 0.f;
    for (int i = tid; i < NROW; i += 256) s += rowloss[i];
    red[tid] = s;
    __syncthreads();
    for (int k = 128; k > 0; k >>= 1) {
        if (tid < k) red[tid] += red[tid + k];
        __syncthreads();
    }
    if (tid == 0) out[0] = red[0] / (float)NROW;
}

extern "C" void kernel_launch(void* const* d_in, const int* in_sizes, int n_in,
                              void* d_out, int out_size, void* d_ws, size_t ws_size,
                              hipStream_t stream) {
    const float* f1 = (const float*)d_in[0];
    const float* f2 = (const float*)d_in[1];

    char* ws = (char*)d_ws;
    _Float16* n1 = (_Float16*)ws;                                    // 6 MB
    _Float16* n2 = (_Float16*)(ws + (size_t)NROW * DDIM * 2);        // 6 MB
    char* p = ws + (size_t)2 * NROW * DDIM * 2;
    float* partial = (float*)p;                                      // 8*4096*4 = 128 KB
    float* diag    = (float*)(p + (size_t)NCHUNK * NROW * 4);        // 16 KB
    float* rowloss = (float*)(p + (size_t)(NCHUNK + 1) * NROW * 4);  // 16 KB

    norm_rows_kernel<<<2 * NROW, 256, 0, stream>>>(f1, f2, n1, n2);

    dim3 grid(NROW / 128, NCHUNK);
    size_t ldsBytes = 2 * 16 * DDIM * sizeof(_Float16);              // 48 KB
    contrastive_gemm_kernel<<<grid, 256, ldsBytes, stream>>>(n1, n2, partial, diag);

    combine_kernel<<<NROW / 256, 256, 0, stream>>>(partial, diag, rowloss);
    reduce_loss_kernel<<<1, 256, 0, stream>>>(rowloss, (float*)d_out);
}
